// GeometricBilinear_40630390620807
// MI455X (gfx1250) — matrix operand order repaired and checked by reading, not compiled, not run on hardware
//
#include <hip/hip_runtime.h>
#include <stdint.h>

typedef __attribute__((ext_vector_type(16))) __bf16 bf16x16;
typedef __attribute__((ext_vector_type(8)))  __bf16 bf16x8;
typedef __attribute__((ext_vector_type(8)))  float  f32x8;
typedef __attribute__((ext_vector_type(4)))  int    i32x4;

#define AS1 __attribute__((address_space(1)))
#define AS3 __attribute__((address_space(3)))

#if __has_builtin(__builtin_amdgcn_global_load_async_to_lds_b128)
#define HAVE_ASYNC_LDS 1
#else
#define HAVE_ASYNC_LDS 0
#endif

__device__ __forceinline__ void wait_asynccnt0() {
#if __has_builtin(__builtin_amdgcn_s_wait_asynccnt)
  __builtin_amdgcn_s_wait_asynccnt(0);
#else
  asm volatile("s_wait_asynccnt 0x0" ::: "memory");
#endif
}

// ---------------- compile-time geometric-algebra tables ----------------
namespace ga {
constexpr int kBits[16] = {0,1,2,4,8,3,5,9,6,10,12,7,11,13,14,15};
constexpr int b2i(int b){ int r=0; for(int t=0;t<16;++t) if(kBits[t]==b) r=t; return r; }
struct CayleyTab { signed char i[192], j[192], k[192], s[192]; };
constexpr CayleyTab makeCayley(){
  CayleyTab T{}; int n=0;
  for(int i=0;i<16;++i) for(int j=0;j<16;++j){
    int a=kBits[i], b=kBits[j];
    if(a&b&1) continue;                       // both contain e0 -> zero
    int sw=0, ash=a>>1;
    while(ash){ int x=ash&b; while(x){ sw += (x&1); x >>= 1; } ash >>= 1; }
    T.i[n]=(signed char)i; T.j[n]=(signed char)j;
    T.k[n]=(signed char)b2i(a^b); T.s[n]=(signed char)((sw&1)?-1:1); ++n;
  }
  return T;
}
constexpr CayleyTab kCayley = makeCayley();
constexpr int  kGrade[16] = {0,1,1,1,1,2,2,2,2,2,2,3,3,3,3,4};
constexpr int  kE0Src[16] = {-1,0,-1,-1,-1,2,3,4,-1,-1,-1,8,9,10,-1,14};
constexpr int  kE0W[16]   = { 0,5, 0, 0, 0,6,6,6, 0, 0, 0,7,7, 7, 0,8};
constexpr bool kNonE0[16] = {true,false,true,true,true,false,false,false,
                             true,true,true,false,false,false,true,false};
} // namespace ga

// ---------------- WMMA helpers (CDNA5 wave32, 16x16x32 bf16) ----------------
__device__ __forceinline__ f32x8 wmma_bf16(bf16x16 a, bf16x16 b, f32x8 c){
  return __builtin_amdgcn_wmma_f32_16x16x32_bf16(false, a, false, b, (short)0, c, false, false);
}
// A fragment 16x32 from a row-major [16][ld] bf16 tile (tokens x K).
__device__ __forceinline__ bf16x16 frag_a(const __bf16* tile, int ld, int k0){
  const int lane = threadIdx.x & 31;
  const int row  = lane & 15;
  const int koff = (lane >> 4) * 8;
  const __bf16* p = tile + row*ld + k0 + koff;
  union { bf16x16 v; bf16x8 h[2]; } u;
  u.h[0] = *(const bf16x8*)(p);
  u.h[1] = *(const bf16x8*)(p + 16);
  return u.v;
}
// B fragment 32x16 from a [n][k] bf16 panel (k contiguous, row stride ldk).
__device__ __forceinline__ bf16x16 frag_b(const __bf16* B, int n0, int k0, int ldk){
  const int lane = threadIdx.x & 31;
  const int col  = lane & 15;
  const int kh   = (lane >> 4) * 8;
  const __bf16* p = B + (size_t)(n0 + col)*ldk + k0 + kh;
  union { bf16x16 v; bf16x8 h[2]; } u;
  u.h[0] = *(const bf16x8*)(p);
  u.h[1] = *(const bf16x8*)(p + 16);
  return u.v;
}

// ---------------- kernel 0: weight repack fp32 -> bf16 GEMM panels ----------------
__global__ void k_repack(const float* __restrict__ wL, const float* __restrict__ wR,
                         const float* __restrict__ wO,
                         const float* __restrict__ s2L, const float* __restrict__ s2R,
                         const float* __restrict__ s2O,
                         const float* __restrict__ m2s, const float* __restrict__ s2sM,
                         __bf16* pWL, __bf16* pWR, __bf16* pWO,
                         __bf16* pS2L, __bf16* pS2R, __bf16* pS2O,
                         __bf16* pM2S, __bf16* pS2S){
  const int W_TASKS = 3*384*384;     // (set, o, i): each copies 9 grade planes
  const int S2_E    = 384*128;
  const int M2S_E   = 128*384;
  const int S2S_E   = 128*128;
  const int total   = W_TASKS + 3*S2_E + M2S_E + S2S_E;
  for (int t = blockIdx.x*blockDim.x + threadIdx.x; t < total; t += gridDim.x*blockDim.x) {
    if (t < W_TASKS) {
      const int set = t / (384*384); const int oi = t % (384*384);
      const float* w = (set==0)?wL:(set==1)?wR:wO;
      __bf16* p      = (set==0)?pWL:(set==1)?pWR:pWO;
      #pragma unroll
      for (int g = 0; g < 9; ++g)
        p[(size_t)g*384*384 + oi] = (__bf16)w[(size_t)oi*9 + g];  // panel[g][o][i]
    } else if (t < W_TASKS + 3*S2_E) {
      const int u = t - W_TASKS; const int set = u / S2_E; const int e = u % S2_E;
      const float* s = (set==0)?s2L:(set==1)?s2R:s2O;
      __bf16* p      = (set==0)?pS2L:(set==1)?pS2R:pS2O;
      p[e] = (__bf16)s[e];                                        // already [o][s]
    } else if (t < W_TASKS + 3*S2_E + M2S_E) {
      const int e = t - W_TASKS - 3*S2_E; pM2S[e] = (__bf16)m2s[e];
    } else {
      const int e = t - W_TASKS - 3*S2_E - M2S_E; pS2S[e] = (__bf16)s2sM[e];
    }
  }
}

// ---------------- kernel 1: left/right equi-linear + geometric product ----------------
// gp_ws is written COMP-MAJOR: [comp][4096 tok][384 ch] bf16 (planes of 3 MB)
__global__ __launch_bounds__(256)
void k_bilinear(const float* __restrict__ mv, const float* __restrict__ sc,
                const __bf16* __restrict__ pWL, const __bf16* __restrict__ pWR,
                const __bf16* __restrict__ pS2L, const __bf16* __restrict__ pS2R,
                const float* __restrict__ bL, const float* __restrict__ bR,
                __bf16* __restrict__ gp_ws){
  extern __shared__ char smem[];
  __bf16* mv_lds = (__bf16*)smem;                              // [16 comp][16 t][384]
  __bf16* sc_lds = (__bf16*)(smem + (size_t)16*16*384*2);      // [16 t][128]
  __bf16* lr_lds = (__bf16*)(smem + (size_t)16*16*384*2 + 16*128*2); // [2][16][16][96]
  const int tid = threadIdx.x;
  const int lane = tid & 31, wave = tid >> 5;
  const int tok0 = blockIdx.x * 16;

  // stage multivectors: fp32 [t][i][16comp] -> bf16 [comp][t][i]
  for (int idx = tid; idx < 16*384*4; idx += 256) {
    const int c4 = idx & 3, rem = idx >> 2;
    const int i = rem % 384, t = rem / 384;
    const float4 v = *(const float4*)(mv + ((size_t)(tok0 + t)*384 + i)*16 + c4*4);
    mv_lds[((c4*4+0)*16 + t)*384 + i] = (__bf16)v.x;
    mv_lds[((c4*4+1)*16 + t)*384 + i] = (__bf16)v.y;
    mv_lds[((c4*4+2)*16 + t)*384 + i] = (__bf16)v.z;
    mv_lds[((c4*4+3)*16 + t)*384 + i] = (__bf16)v.w;
  }
  for (int idx = tid; idx < 16*128; idx += 256) {
    const int t = idx >> 7, s = idx & 127;
    sc_lds[idx] = (__bf16)sc[(size_t)(tok0 + t)*128 + s];
  }
  __syncthreads();

  for (int chunk = 0; chunk < 4; ++chunk) {
    // ---- GEMM phase: 192 blocks = 2 sides x 16 comps x 6 n-tiles of 16 ----
    for (int blk = wave; blk < 192; blk += 8) {
      const int side = blk / 96, rem = blk % 96;
      const int comp = rem / 6,  nt  = rem % 6;
      const int n0g  = chunk*96 + nt*16;
      const __bf16* W    = side ? pWR  : pWL;
      const __bf16* S2   = side ? pS2R : pS2L;
      const float*  bias = side ? bR   : bL;
      const __bf16* At = mv_lds + (size_t)comp*16*384;
      const __bf16* Bg = W + (size_t)ga::kGrade[comp]*384*384;
      __builtin_prefetch(Bg + (size_t)n0g*384, 0, 1);
      f32x8 acc = {0.f,0.f,0.f,0.f,0.f,0.f,0.f,0.f};
      #pragma unroll 4
      for (int k = 0; k < 384; k += 32)
        acc = wmma_bf16(frag_a(At, 384, k), frag_b(Bg, n0g, k, 384), acc);
      if (ga::kE0Src[comp] >= 0) {            // e0 cross-component term
        const __bf16* As = mv_lds + (size_t)ga::kE0Src[comp]*16*384;
        const __bf16* Bh = W + (size_t)ga::kE0W[comp]*384*384;
        __builtin_prefetch(Bh + (size_t)n0g*384, 0, 1);
        #pragma unroll 4
        for (int k = 0; k < 384; k += 32)
          acc = wmma_bf16(frag_a(As, 384, k), frag_b(Bh, n0g, k, 384), acc);
      }
      if (comp == 0) {                        // scalar injection + bias
        #pragma unroll
        for (int k = 0; k < 128; k += 32)
          acc = wmma_bf16(frag_a(sc_lds, 128, k), frag_b(S2, n0g, k, 128), acc);
        const float bv = bias[n0g + (lane & 15)];
        #pragma unroll
        for (int r = 0; r < 8; ++r) acc[r] += bv;
      }
      __bf16* dst = lr_lds + (size_t)(side*16 + comp)*16*96;
      const int col = nt*16 + (lane & 15);
      #pragma unroll
      for (int r = 0; r < 8; ++r) {
        const int t = r + ((lane >> 4) << 3);
        dst[t*96 + col] = (__bf16)acc[r];
      }
    }
    __syncthreads();
    // ---- sparse Cayley geometric product on this 96-channel chunk ----
    for (int p = tid; p < 16*96; p += 256) {
      const int t = p / 96, o = p % 96;
      float L[16], R[16], G[16];
      #pragma unroll
      for (int c = 0; c < 16; ++c) {
        L[c] = (float)lr_lds[(c*16 + t)*96 + o];
        R[c] = (float)lr_lds[((16 + c)*16 + t)*96 + o];
        G[c] = 0.f;
      }
      #pragma unroll
      for (int n = 0; n < 192; ++n) {
        const float pr = L[ga::kCayley.i[n]] * R[ga::kCayley.j[n]];
        G[ga::kCayley.k[n]] += (ga::kCayley.s[n] > 0) ? pr : -pr;
      }
      // comp-major store: consecutive lanes -> consecutive channels (coalesced)
      #pragma unroll
      for (int c = 0; c < 16; ++c)
        gp_ws[((size_t)c*4096 + tok0 + t)*384 + chunk*96 + o] = (__bf16)G[c];
    }
    __syncthreads();
  }
}

// ---------------- kernel 2: output equi-linear + scalar head (un-normalized) ----------------
__global__ __launch_bounds__(256)
void k_output(const __bf16* __restrict__ gp_ws, const float* __restrict__ sc,
              const __bf16* __restrict__ pWO, const __bf16* __restrict__ pS2O,
              const float* __restrict__ bO,
              const __bf16* __restrict__ pM2S, const __bf16* __restrict__ pS2S,
              const float* __restrict__ bS,
              float* __restrict__ out_mv, float* __restrict__ out_s,
              float* __restrict__ sumsq_ws){
  extern __shared__ char smem[];
  __bf16* gp_lds = (__bf16*)smem;                              // [16 comp][16 t][384]
  __bf16* sc_lds = (__bf16*)(smem + (size_t)16*16*384*2);      // [16 t][128]
  float*  ssq    = (float*)(smem + (size_t)16*16*384*2 + 16*128*2); // [16]
  const int tid = threadIdx.x;
  const int lane = tid & 31, wave = tid >> 5;
  const int tok0 = blockIdx.x * 16;

  if (tid < 16) ssq[tid] = 0.f;
  // stage gp tile: comp-major global -> identical comp-major LDS (16B chunks).
  // CDNA5 async copy path when available (ASYNCcnt-tracked, no VGPR round-trip).
  for (int idx = tid; idx < 16*16*48; idx += 256) {   // 12288 x 16B
    const int comp = idx / 768;          // 16 t * 48 chunks
    const int rem  = idx % 768;
    const int t    = rem / 48;
    const int h8   = (rem % 48) * 8;
    const __bf16* src = gp_ws + ((size_t)comp*4096 + tok0 + t)*384 + h8;
    __bf16*       dst = gp_lds + (comp*16 + t)*384 + h8;
#if HAVE_ASYNC_LDS
    __builtin_amdgcn_global_load_async_to_lds_b128(
        (AS1 i32x4*)src, (AS3 i32x4*)dst, 0, 0);
#else
    *(uint4*)dst = *(const uint4*)src;
#endif
  }
  for (int idx = tid; idx < 16*128; idx += 256) {
    const int t = idx >> 7, s = idx & 127;
    sc_lds[idx] = (__bf16)sc[(size_t)(tok0 + t)*128 + s];
  }
#if HAVE_ASYNC_LDS
  wait_asynccnt0();
#endif
  __syncthreads();

  // 384 mv blocks (16 comps x 24 n-tiles) + 8 scalar-head blocks
  for (int blk = wave; blk < 392; blk += 8) {
    if (blk < 384) {
      const int comp = blk / 24, nt = blk % 24;
      const int n0 = nt * 16;
      const __bf16* At = gp_lds + (size_t)comp*16*384;
      const __bf16* Bg = pWO + (size_t)ga::kGrade[comp]*384*384;
      __builtin_prefetch(Bg + (size_t)n0*384, 0, 1);
      f32x8 acc = {0.f,0.f,0.f,0.f,0.f,0.f,0.f,0.f};
      #pragma unroll 4
      for (int k = 0; k < 384; k += 32)
        acc = wmma_bf16(frag_a(At, 384, k), frag_b(Bg, n0, k, 384), acc);
      if (ga::kE0Src[comp] >= 0) {
        const __bf16* As = gp_lds + (size_t)ga::kE0Src[comp]*16*384;
        const __bf16* Bh = pWO + (size_t)ga::kE0W[comp]*384*384;
        #pragma unroll 4
        for (int k = 0; k < 384; k += 32)
          acc = wmma_bf16(frag_a(As, 384, k), frag_b(Bh, n0, k, 384), acc);
      }
      if (comp == 0) {
        #pragma unroll
        for (int k = 0; k < 128; k += 32)
          acc = wmma_bf16(frag_a(sc_lds, 128, k), frag_b(pS2O, n0, k, 128), acc);
        const float bv = bO[n0 + (lane & 15)];
        #pragma unroll
        for (int r = 0; r < 8; ++r) acc[r] += bv;
      }
      const int col = n0 + (lane & 15);
      #pragma unroll
      for (int r = 0; r < 8; ++r) {
        const int t = r + ((lane >> 4) << 3);
        out_mv[((size_t)(tok0 + t)*384 + col)*16 + comp] = acc[r];
      }
      if (ga::kNonE0[comp]) {                 // accumulate per-token sum of squares
        #pragma unroll
        for (int r = 0; r < 8; ++r) {
          float v = acc[r]*acc[r];
          v += __shfl_xor(v, 1); v += __shfl_xor(v, 2);
          v += __shfl_xor(v, 4); v += __shfl_xor(v, 8);
          if ((lane & 15) == 0) atomicAdd(&ssq[r + ((lane >> 4) << 3)], v);
        }
      }
    } else {                                  // scalar head: gp0@mvs2s^T + sc@s2s^T + b
      const int n0 = (blk - 384) * 16;
      f32x8 acc = {0.f,0.f,0.f,0.f,0.f,0.f,0.f,0.f};
      #pragma unroll 4
      for (int k = 0; k < 384; k += 32)
        acc = wmma_bf16(frag_a(gp_lds, 384, k), frag_b(pM2S, n0, k, 384), acc);
      #pragma unroll
      for (int k = 0; k < 128; k += 32)
        acc = wmma_bf16(frag_a(sc_lds, 128, k), frag_b(pS2S, n0, k, 128), acc);
      const float bv = bS[n0 + (lane & 15)];
      const int col = n0 + (lane & 15);
      #pragma unroll
      for (int r = 0; r < 8; ++r) {
        const int t = r + ((lane >> 4) << 3);
        out_s[(size_t)(tok0 + t)*128 + col] = acc[r] + bv;
      }
    }
  }
  __syncthreads();
  if (tid < 16) sumsq_ws[tok0 + tid] = ssq[tid];
}

// ---------------- kernel 3: mv scaling + scalar layer-norm ----------------
__global__ __launch_bounds__(256)
void k_norm(float* __restrict__ out_mv, float* __restrict__ out_s,
            const float* __restrict__ sumsq_ws){
  __shared__ float sval[128];
  __shared__ float red[2];
  const int t = blockIdx.x;
  const int tid = threadIdx.x;
  const float nf = rsqrtf(sumsq_ws[t] * (1.f/384.f) + 0.01f);
  float4* row = (float4*)(out_mv + (size_t)t * 6144);
  #pragma unroll
  for (int j = 0; j < 6; ++j) {
    float4 q = row[tid + 256*j];
    q.x *= nf; q.y *= nf; q.z *= nf; q.w *= nf;
    row[tid + 256*j] = q;
  }
  if (tid == 0) { red[0] = 0.f; red[1] = 0.f; }
  __syncthreads();
  float* srow = out_s + (size_t)t * 128;
  if (tid < 128) {
    const float v = srow[tid]; sval[tid] = v;
    atomicAdd(&red[0], v); atomicAdd(&red[1], v*v);
  }
  __syncthreads();
  if (tid < 128) {
    const float m   = red[0] * (1.f/128.f);
    const float var = red[1] * (1.f/128.f) - m*m;
    srow[tid] = (sval[tid] - m) * rsqrtf(var + 1e-5f);
  }
}

// ---------------- host ----------------
extern "C" void kernel_launch(void* const* d_in, const int* in_sizes, int n_in,
                              void* d_out, int out_size, void* d_ws, size_t ws_size,
                              hipStream_t stream) {
  (void)in_sizes; (void)n_in; (void)out_size; (void)ws_size;
  const float* mv  = (const float*)d_in[0];
  const float* sc  = (const float*)d_in[1];
  const float* wL  = (const float*)d_in[2];
  const float* s2L = (const float*)d_in[3];
  const float* bL  = (const float*)d_in[4];
  const float* wR  = (const float*)d_in[5];
  const float* s2R = (const float*)d_in[6];
  const float* bR  = (const float*)d_in[7];
  const float* wO  = (const float*)d_in[8];
  const float* s2O = (const float*)d_in[9];
  const float* bO  = (const float*)d_in[10];
  const float* m2s = (const float*)d_in[11];
  const float* s2s = (const float*)d_in[12];
  const float* bS  = (const float*)d_in[13];

  char* ws = (char*)d_ws;
  size_t off = 0;
  auto take = [&](size_t bytes) { char* p = ws + off; off += (bytes + 255) & ~(size_t)255; return p; };
  __bf16* pWL  = (__bf16*)take((size_t)9*384*384*2);
  __bf16* pWR  = (__bf16*)take((size_t)9*384*384*2);
  __bf16* pWO  = (__bf16*)take((size_t)9*384*384*2);
  __bf16* pS2L = (__bf16*)take((size_t)384*128*2);
  __bf16* pS2R = (__bf16*)take((size_t)384*128*2);
  __bf16* pS2O = (__bf16*)take((size_t)384*128*2);
  __bf16* pM2S = (__bf16*)take((size_t)128*384*2);
  __bf16* pS2S = (__bf16*)take((size_t)128*128*2);
  __bf16* gp   = (__bf16*)take((size_t)4096*384*16*2);
  float*  ssq  = (float*) take((size_t)4096*4);

  float* out_mv = (float*)d_out;
  float* out_s  = out_mv + (size_t)4096*384*16;

  const uint32_t SMEM1 = (16*16*384 + 16*128 + 2*16*16*96) * 2;          // 299008 B
  const uint32_t SMEM2 = (16*16*384 + 16*128) * 2 + 16*sizeof(float);    // 200768 B

  k_repack<<<2560, 256, 0, stream>>>(wL, wR, wO, s2L, s2R, s2O, m2s, s2s,
                                     pWL, pWR, pWO, pS2L, pS2R, pS2O, pM2S, pS2S);
  k_bilinear<<<256, 256, SMEM1, stream>>>(mv, sc, pWL, pWR, pS2L, pS2R, bL, bR, gp);
  k_output<<<256, 256, SMEM2, stream>>>(gp, sc, pWO, pS2O, bO, pM2S, pS2S, bS,
                                        out_mv, out_s, ssq);
  k_norm<<<4096, 256, 0, stream>>>(out_mv, out_s, ssq);
}
